// GatedResidualHeadAssociativeLM_34368328303124
// MI455X (gfx1250) — compile-verified, matrix operand order checked by reading, not run on hardware
//
#include <hip/hip_runtime.h>
#include <hip/hip_bf16.h>

// Problem dims (fixed by the reference)
#define KV 32000
#define KE 512
#define KH 1024
#define KM 256
#define KB 2
#define KS 1024
#define KMS (KB * KS) // 2048 total rows (B*S)

typedef __attribute__((ext_vector_type(16))) __bf16 v16bf;
typedef __attribute__((ext_vector_type(8)))  float  v8f;

struct __attribute__((aligned(16))) U128 { unsigned int x[4]; };
union Frag { v16bf v; U128 q[2]; };

__device__ __forceinline__ unsigned short f32_to_bf16(float f) {
  union { float f; unsigned int u; } c; c.f = f;
  unsigned int u = c.u + 0x7FFFu + ((c.u >> 16) & 1u); // RNE
  return (unsigned short)(u >> 16);
}

// CDNA5 async copy: global -> LDS, 16 bytes per lane, tracked by ASYNCcnt.
// (CDNA5 ISA: GLOBAL_LOAD_ASYNC_TO_LDS_B128, GV mode.)
__device__ __forceinline__ void async_g2l_b128(unsigned lds_byte_addr,
                                               const void* gaddr) {
  asm volatile("global_load_async_to_lds_b128 %0, %1, off"
               :: "v"(lds_byte_addr), "v"(gaddr)
               : "memory");
}
__device__ __forceinline__ void wait_asynccnt0() {
  asm volatile("s_wait_asynccnt 0" ::: "memory");
}
// Flat LDS pointer -> LDS byte address (ISA §10.2: LDS aperture uses addr[31:0]).
__device__ __forceinline__ unsigned lds_addr_of(const void* p) {
  return (unsigned)(uintptr_t)p;
}

// Load a 16-element bf16 fragment for one lane from a K-contiguous row.
// ISA 16-bit A layout: lanes 0-15 hold K{0..7,16..23}, lanes 16-31 hold K{8..15,24..31}.
__device__ __forceinline__ v16bf frag_from(const unsigned short* rowp) {
  const int lane = threadIdx.x & 31;
  const int half = (lane >> 4) << 3; // 0 or 8
  Frag f;
  f.q[0] = *reinterpret_cast<const U128*>(rowp + half);
  f.q[1] = *reinterpret_cast<const U128*>(rowp + 16 + half);
  return f.v;
}

// ---------------------------------------------------------------------------
// bf16 WMMA GEMM with double-buffered async-LDS staging:
//   C[m,n] = act( sum_k A[m,k]*B[n,k] + bias[n] )
// A: [M,K] bf16 bits, B: [N,K] bf16 bits (X @ W.T with row-major W).
// Block = 256 threads = 8 waves; wave -> 16x64 tile; block -> 128x64.
// Per K-step the block async-copies A[128x32] (8KB) + B[64x32] (4KB) to LDS;
// B is fetched from global once per block instead of once per wave.
// Requires M%128==0, N%64==0, K%32==0 (true for all calls here).
// ---------------------------------------------------------------------------
__global__ __launch_bounds__(256) void gemm_bf16_kernel(
    const unsigned short* __restrict__ A,
    const unsigned short* __restrict__ B,
    const float* __restrict__ bias,
    float* __restrict__ Cf,
    unsigned short* __restrict__ Cb,
    int Mdim, int Ndim, int Kdim, int act)
{
  __shared__ __align__(16) unsigned short ldsA[2][128 * 32];
  __shared__ __align__(16) unsigned short ldsB[2][64 * 32];

  const int tid  = threadIdx.x;
  const int wave = tid >> 5;
  const int lane = tid & 31;
  const int m0 = blockIdx.y * 128;
  const int n0 = blockIdx.x * 64;

  // Per-thread async-copy slots (16B chunks; chunk c -> row c>>2, k-offset (c&3)*8)
  const int arow0 = tid >> 1;                 // A chunk 2*tid   : rows 0..127
  const int ak0   = (tid & 1) * 16;           // two 16B chunks per row half
  const int brow  = tid >> 2;                 // B chunk tid     : rows 0..63
  const int bk0   = (tid & 3) * 8;

  auto issue_tile = [&](int buf, int k0) {
    // A tile: 512 x 16B, 2 chunks per thread
    {
      const unsigned short* g = A + (size_t)(m0 + arow0) * Kdim + k0 + ak0;
      async_g2l_b128(lds_addr_of(&ldsA[buf][arow0 * 32 + ak0]), g);
      async_g2l_b128(lds_addr_of(&ldsA[buf][arow0 * 32 + ak0 + 8]), g + 8);
    }
    // B tile: 256 x 16B, 1 chunk per thread
    {
      const unsigned short* g = B + (size_t)(n0 + brow) * Kdim + k0 + bk0;
      async_g2l_b128(lds_addr_of(&ldsB[buf][brow * 32 + bk0]), g);
    }
  };

  v8f acc[4] = {v8f{}, v8f{}, v8f{}, v8f{}};

  const int nk = Kdim / 32;
  issue_tile(0, 0);
  for (int i = 0; i < nk; ++i) {
    const int cur = i & 1;
    wait_asynccnt0();      // my copies for buf `cur` have landed
    __syncthreads();       // everyone's copies landed; prev buffer fully consumed
    if (i + 1 < nk) issue_tile(cur ^ 1, (i + 1) * 32);

    v16bf afrag = frag_from(&ldsA[cur][(wave * 16 + (lane & 15)) * 32]);
    v16bf bfrag[4];
#pragma unroll
    for (int nn = 0; nn < 4; ++nn)
      bfrag[nn] = frag_from(&ldsB[cur][(nn * 16 + (lane & 15)) * 32]);
#pragma unroll
    for (int nn = 0; nn < 4; ++nn)
      acc[nn] = __builtin_amdgcn_wmma_f32_16x16x32_bf16(
          false, afrag, false, bfrag[nn], (short)0, acc[nn], false, false);
    __syncthreads();       // done reading buf `cur` before it is overwritten
  }

  // Epilogue. D layout: VGPR v -> M = v (lanes 0-15) or v+8 (lanes 16-31); N = lane%16.
  const int mw = m0 + wave * 16;
  const int half8 = (lane >> 4) << 3;
  const int colb = n0 + (lane & 15);
#pragma unroll
  for (int nn = 0; nn < 4; ++nn) {
    const int col = colb + nn * 16;
    const float bv = bias ? bias[col] : 0.0f;
#pragma unroll
    for (int v = 0; v < 8; ++v) {
      const int row = mw + v + half8;
      float x = acc[nn][v] + bv;
      if (act == 1) { x = fmaxf(x, 0.0f); x = x * x; }
      else if (act == 2) { x = fmaxf(x, 0.0f); }
      const size_t idx = (size_t)row * Ndim + col;
      if (Cf) Cf[idx] = x;
      if (Cb) Cb[idx] = f32_to_bf16(x);
    }
  }
}

// ---------------------------------------------------------------------------
// Elementwise f32 -> bf16 bits
// ---------------------------------------------------------------------------
__global__ void cvt_bf16_kernel(const float* __restrict__ in,
                                unsigned short* __restrict__ out, int n) {
  int i = blockIdx.x * blockDim.x + threadIdx.x;
  if (i < n) out[i] = f32_to_bf16(in[i]);
}

// Embedding gather straight into bf16 activations: emb_bf[m,:] = embW_bf[ids[m],:]
__global__ void gather_kernel(const int* __restrict__ ids,
                              const unsigned short* __restrict__ embW_bf,
                              unsigned short* __restrict__ emb_bf) {
  const int m = blockIdx.x;          // 0..KMS-1
  const int id = ids[m];
  for (int e = threadIdx.x; e < KE; e += blockDim.x)
    emb_bf[(size_t)m * KE + e] = embW_bf[(size_t)id * KE + e];
}

__global__ void zero_kernel(float* __restrict__ p, int n) {
  int i = blockIdx.x * blockDim.x + threadIdx.x;
  if (i < n) p[i] = 0.0f;
}

// ---------------------------------------------------------------------------
// One GRU timestep. One wave32 per (batch, hidden unit): three K=1024 dots
// (r,z,n rows of w_hh) with shuffle reduction, then the gate update.
// h double-buffered across launches (h_in -> h_out).
// ---------------------------------------------------------------------------
__global__ __launch_bounds__(256) void gru_step_kernel(
    const float* __restrict__ xp,     // [KMS, 3H] (= x@w_ih.T + b_ih)
    const float* __restrict__ w_hh,   // [3H, H]
    const float* __restrict__ b_hh,   // [3H]
    const float* __restrict__ h_in,   // [B, H]
    float* __restrict__ h_out,        // [B, H]
    float* __restrict__ states,       // [KMS, H]
    int t)
{
  const int gw   = (blockIdx.x * blockDim.x + threadIdx.x) >> 5; // 0..2047
  const int lane = threadIdx.x & 31;
  const int b = gw >> 10;       // /KH
  const int j = gw & (KH - 1);

  const float* hb = h_in + b * KH;
  const float* wr = w_hh + (size_t)j * KH;
  const float* wz = w_hh + (size_t)(KH + j) * KH;
  const float* wn = w_hh + (size_t)(2 * KH + j) * KH;

  float sr = 0.f, sz = 0.f, sn = 0.f;
  for (int k = lane; k < KH; k += 32) {
    const float hv = hb[k];
    sr = fmaf(wr[k], hv, sr);
    sz = fmaf(wz[k], hv, sz);
    sn = fmaf(wn[k], hv, sn);
  }
  for (int off = 16; off; off >>= 1) {
    sr += __shfl_down(sr, off, 32);
    sz += __shfl_down(sz, off, 32);
    sn += __shfl_down(sn, off, 32);
  }
  if (lane == 0) {
    const float* x = xp + ((size_t)b * KS + t) * (3 * KH);
    const float ir = x[j], iz = x[KH + j], in = x[2 * KH + j];
    const float hr = sr + b_hh[j];
    const float hz = sz + b_hh[KH + j];
    const float hn = sn + b_hh[2 * KH + j];
    const float r = 1.0f / (1.0f + __expf(-(ir + hr)));
    const float z = 1.0f / (1.0f + __expf(-(iz + hz)));
    const float n = tanhf(in + r * hn);
    const float h = (1.0f - z) * n + z * hb[j];
    h_out[b * KH + j] = h;
    states[((size_t)b * KS + t) * KH + j] = h;
  }
}

// gate[m] = sigmoid(states[m,:].Wg + bg). One wave per row.
__global__ __launch_bounds__(256) void gate_kernel(
    const float* __restrict__ states, const float* __restrict__ Wg,
    const float* __restrict__ bg, float* __restrict__ gate) {
  const int gw   = (blockIdx.x * blockDim.x + threadIdx.x) >> 5; // 0..2047
  const int lane = threadIdx.x & 31;
  const float* s = states + (size_t)gw * KH;
  float acc = 0.f;
  for (int k = lane; k < KH; k += 32) acc = fmaf(s[k], Wg[k], acc);
  for (int off = 16; off; off >>= 1) acc += __shfl_down(acc, off, 32);
  if (lane == 0) gate[gw] = 1.0f / (1.0f + __expf(-(acc + bg[0])));
}

// ---------------------------------------------------------------------------
// Strictly-causal attention + gating. One block (256 thr) per query row.
// gated[b,i,j] = softmax_{j<i}(q_i.k_j/16) * gate[b,i]*memory_scale, 0 for j>=i.
// ---------------------------------------------------------------------------
__global__ __launch_bounds__(256) void attn_kernel(
    const float* __restrict__ q, const float* __restrict__ kmat,
    const float* __restrict__ gate, const float* __restrict__ mscale,
    float* __restrict__ gated) {
  __shared__ float qs[KM];
  __shared__ float red[256];
  const int row = blockIdx.x;           // b*S + i
  const int b = row >> 10;
  const int i = row & (KS - 1);
  const int tid = threadIdx.x;

  qs[tid] = q[(size_t)row * KM + tid];
  __syncthreads();

  float sc[4];
  float mx = -3.4e38f;
#pragma unroll
  for (int jj = 0; jj < 4; ++jj) {
    const int j = tid + jj * 256;
    float s = -3.4e38f;
    if (j < i) {
      const float* kr = kmat + ((size_t)b * KS + j) * KM;
      float d = 0.f;
      for (int e = 0; e < KM; ++e) d = fmaf(qs[e], kr[e], d);
      s = d * 0.0625f; // 1/sqrt(256)
    }
    sc[jj] = s;
    mx = fmaxf(mx, s);
  }
  red[tid] = mx; __syncthreads();
  for (int off = 128; off; off >>= 1) {
    if (tid < off) red[tid] = fmaxf(red[tid], red[tid + off]);
    __syncthreads();
  }
  mx = red[0]; __syncthreads();

  float ex[4], sum = 0.f;
#pragma unroll
  for (int jj = 0; jj < 4; ++jj) {
    const int j = tid + jj * 256;
    const float e = (j < i) ? __expf(sc[jj] - mx) : 0.0f;
    ex[jj] = e; sum += e;
  }
  red[tid] = sum; __syncthreads();
  for (int off = 128; off; off >>= 1) {
    if (tid < off) red[tid] += red[tid + off];
    __syncthreads();
  }
  sum = red[0];

  const float inv = 1.0f / fmaxf(sum, 1e-6f);
  const float g = gate[row] * mscale[0] * inv;
#pragma unroll
  for (int jj = 0; jj < 4; ++jj) {
    const int j = tid + jj * 256;
    gated[(size_t)row * KS + j] = ex[jj] * g;
  }
}

// Acomb[m,k] = base[m,k] + gate[m]*residual[m,k]  (bf16, feeds fused logits GEMM)
__global__ void acomb_kernel(const float* __restrict__ base,
                             const float* __restrict__ residual,
                             const float* __restrict__ gate,
                             unsigned short* __restrict__ out) {
  const size_t idx = (size_t)blockIdx.x * blockDim.x + threadIdx.x; // < KMS*KE
  const int m = (int)(idx >> 9); // /KE
  out[idx] = f32_to_bf16(base[idx] + gate[m] * residual[idx]);
}

// Scatter-add gated attention into logits by key-token id.
__global__ void scatter_kernel(const int* __restrict__ ids,
                               const float* __restrict__ gated,
                               float* __restrict__ out) {
  const size_t idx = (size_t)blockIdx.x * blockDim.x + threadIdx.x; // < KB*KS*KS
  const int j = (int)(idx & (KS - 1));
  const int row = (int)(idx >> 10);  // b*S + i
  const int i = row & (KS - 1);
  const int b = row >> 10;
  if (j >= i) return;
  const float g = gated[idx];
  const int tok = ids[b * KS + j];
  atomicAdd(out + (size_t)row * KV + tok, g);
}

// ---------------------------------------------------------------------------
extern "C" void kernel_launch(void* const* d_in, const int* in_sizes, int n_in,
                              void* d_out, int out_size, void* d_ws, size_t ws_size,
                              hipStream_t stream) {
  const int*   input_ids = (const int*)  d_in[0];
  const float* emb_W     = (const float*)d_in[1];
  const float* w_ih      = (const float*)d_in[2];
  const float* w_hh      = (const float*)d_in[3];
  const float* b_ih      = (const float*)d_in[4];
  const float* b_hh      = (const float*)d_in[5];
  const float* Wq        = (const float*)d_in[6];
  const float* bq        = (const float*)d_in[7];
  const float* Wk        = (const float*)d_in[8];
  const float* bk        = (const float*)d_in[9];
  const float* Wg        = (const float*)d_in[10];
  const float* bg        = (const float*)d_in[11];
  const float* W1        = (const float*)d_in[12];
  const float* b1        = (const float*)d_in[13];
  const float* W2        = (const float*)d_in[14];
  const float* b2        = (const float*)d_in[15];
  const float* Wr        = (const float*)d_in[16];
  const float* br        = (const float*)d_in[17];
  const float* mscale    = (const float*)d_in[18];
  const float* out_bias  = (const float*)d_in[19];
  float* out = (float*)d_out;

  // Workspace carve-out
  char* ws = (char*)d_ws;
  size_t off = 0;
  auto alloc = [&](size_t bytes) -> void* {
    off = (off + 255) & ~(size_t)255;
    void* p = ws + off;
    off += bytes;
    return p;
  };
  unsigned short* embW_bf  = (unsigned short*)alloc((size_t)KV * KE * 2);
  unsigned short* emb_bf   = (unsigned short*)alloc((size_t)KMS * KE * 2);
  unsigned short* wih_bf   = (unsigned short*)alloc((size_t)3 * KH * KE * 2);
  float*          xp       = (float*)        alloc((size_t)KMS * 3 * KH * 4);
  float*          hA       = (float*)        alloc((size_t)KB * KH * 4);
  float*          hB       = (float*)        alloc((size_t)KB * KH * 4);
  float*          states   = (float*)        alloc((size_t)KMS * KH * 4);
  unsigned short* states_bf= (unsigned short*)alloc((size_t)KMS * KH * 2);
  unsigned short* W1_bf    = (unsigned short*)alloc((size_t)4 * KE * KH * 2);
  unsigned short* hf_bf    = (unsigned short*)alloc((size_t)KMS * 4 * KE * 2);
  unsigned short* W2_bf    = (unsigned short*)alloc((size_t)KE * 4 * KE * 2);
  float*          base     = (float*)        alloc((size_t)KMS * KE * 4);
  unsigned short* base_bf  = (unsigned short*)alloc((size_t)KMS * KE * 2);
  unsigned short* Wr_bf    = (unsigned short*)alloc((size_t)KE * KE * 2);
  float*          residual = (float*)        alloc((size_t)KMS * KE * 4);
  unsigned short* Wq_bf    = (unsigned short*)alloc((size_t)KM * KH * 2);
  unsigned short* Wk_bf    = (unsigned short*)alloc((size_t)KM * KH * 2);
  float*          qbuf     = (float*)        alloc((size_t)KMS * KM * 4);
  float*          kbuf     = (float*)        alloc((size_t)KMS * KM * 4);
  float*          gate     = (float*)        alloc((size_t)KMS * 4);
  float*          gated    = (float*)        alloc((size_t)KB * KS * KS * 4);
  unsigned short* acomb_bf = (unsigned short*)alloc((size_t)KMS * KE * 2);

  auto cvt = [&](const float* src, unsigned short* dst, int n) {
    cvt_bf16_kernel<<<(n + 255) / 256, 256, 0, stream>>>(src, dst, n);
  };
  auto gemm = [&](const unsigned short* A, const unsigned short* Bm,
                  const float* bias, float* Cf, unsigned short* Cb,
                  int Md, int Nd, int Kd, int act) {
    dim3 grid(Nd / 64, Md / 128);
    gemm_bf16_kernel<<<grid, 256, 0, stream>>>(A, Bm, bias, Cf, Cb, Md, Nd, Kd, act);
  };

  // 1) weight conversions to bf16
  cvt(emb_W, embW_bf, KV * KE);
  cvt(w_ih, wih_bf, 3 * KH * KE);
  cvt(W1, W1_bf, 4 * KE * KH);
  cvt(W2, W2_bf, KE * 4 * KE);
  cvt(Wr, Wr_bf, KE * KE);
  cvt(Wq, Wq_bf, KM * KH);
  cvt(Wk, Wk_bf, KM * KH);

  // 2) embedding gather (bf16)
  gather_kernel<<<KMS, 256, 0, stream>>>(input_ids, embW_bf, emb_bf);

  // 3) xp = emb @ w_ih.T + b_ih   [2048 x 3072], K=512
  gemm(emb_bf, wih_bf, b_ih, xp, nullptr, KMS, 3 * KH, KE, 0);

  // 4) GRU scan (h double-buffered across sequential launches)
  zero_kernel<<<(KB * KH + 255) / 256, 256, 0, stream>>>(hA, KB * KH);
  float* hbuf[2] = {hA, hB};
  for (int t = 0; t < KS; ++t) {
    gru_step_kernel<<<256, 256, 0, stream>>>(xp, w_hh, b_hh,
                                             hbuf[t & 1], hbuf[(t + 1) & 1],
                                             states, t);
  }

  // 5) states -> bf16
  cvt(states, states_bf, KMS * KH);

  // 6) hf = relu(states@W1.T + b1)^2   [2048 x 2048] bf16, K=1024
  gemm(states_bf, W1_bf, b1, nullptr, hf_bf, KMS, 4 * KE, KH, 1);

  // 7) base = hf@W2.T + b2   [2048 x 512], K=2048
  gemm(hf_bf, W2_bf, b2, base, base_bf, KMS, KE, 4 * KE, 0);

  // 8) residual = relu(base@Wr.T + br)   [2048 x 512], K=512
  gemm(base_bf, Wr_bf, br, residual, nullptr, KMS, KE, KE, 2);

  // 9) gate = sigmoid(states@Wg.T + bg)
  gate_kernel<<<256, 256, 0, stream>>>(states, Wg, bg, gate);

  // 10) q/k projections [2048 x 256], K=1024
  gemm(states_bf, Wq_bf, bq, qbuf, nullptr, KMS, KM, KH, 0);
  gemm(states_bf, Wk_bf, bk, kbuf, nullptr, KMS, KM, KH, 0);

  // 11) strictly-causal softmax attention, pre-scaled by gate*memory_scale
  attn_kernel<<<KMS, 256, 0, stream>>>(qbuf, kbuf, gate, mscale, gated);

  // 12) fold gate*residual into base: one vocab GEMM instead of two
  acomb_kernel<<<(KMS * KE) / 256, 256, 0, stream>>>(base, residual, gate, acomb_bf);

  // 13) logits = Acomb @ emb_W.T + output_bias   [2048 x 32000], K=512
  gemm(acomb_bf, embW_bf, out_bias, out, nullptr, KMS, KV, KE, 0);

  // 14) scatter-add gated attention into logits at key-token columns
  scatter_kernel<<<(KB * KS * KS) / 256, 256, 0, stream>>>(input_ids, gated, out);
}
